// CapsuleNetwork_55259049230360
// MI455X (gfx1250) — matrix operand order repaired and checked by reading, not compiled, not run on hardware
//
#include <hip/hip_runtime.h>
#include <hip/hip_fp16.h>
#include <cstdint>
#include <cstddef>

typedef __attribute__((ext_vector_type(16))) _Float16 v16h;
typedef __attribute__((ext_vector_type(8)))  _Float16 v8h;
typedef __attribute__((ext_vector_type(8)))  float    v8f;

#define BATCH 128
#define USE_ASYNC 1   // gfx1250 global_load_async_to_lds_b128 for A-tile staging

// ---------------------------------------------------------------------------
// elementwise helpers
// ---------------------------------------------------------------------------
__global__ void f32_to_f16_kernel(const float* __restrict__ in,
                                  _Float16* __restrict__ out, int n) {
  int i = blockIdx.x * 256 + threadIdx.x;
  if (i < n) out[i] = (_Float16)in[i];
}

__global__ void zero_f32_kernel(float* __restrict__ p, int n) {
  int i = blockIdx.x * 256 + threadIdx.x;
  if (i < n) p[i] = 0.0f;
}

// ---------------------------------------------------------------------------
// conv1: 1 input channel, 9x9, stride 1, 56x56 -> 48x48, 256 out channels.
// K=81 too small for WMMA to pay; direct f32 FMA, relu, store f16.
// ---------------------------------------------------------------------------
__global__ __launch_bounds__(256)
void conv1_kernel(const float* __restrict__ x,      // [B,1,56,56]
                  const float* __restrict__ w,      // [256,1,9,9]
                  const float* __restrict__ bias,   // [256]
                  _Float16* __restrict__ out) {
  int idx = blockIdx.x * 256 + threadIdx.x;         // 128*256*48*48
  int ox = idx % 48;
  int t  = idx / 48;
  int oy = t % 48;  t /= 48;
  int oc = t % 256;
  int b  = t / 256;
  const float* xp = x + (size_t)b * 56 * 56 + oy * 56 + ox;
  const float* wp = w + oc * 81;
  float acc = bias[oc];
#pragma unroll
  for (int r = 0; r < 9; ++r)
#pragma unroll
    for (int s = 0; s < 9; ++s)
      acc += xp[r * 56 + s] * wp[r * 9 + s];
  out[idx] = (_Float16)fmaxf(acc, 0.0f);
}

// ---------------------------------------------------------------------------
// Implicit-GEMM WMMA conv: 9x9, stride 2, 256->256 channels.
//   D[M=256][Npix] = A[M][K=256*81] * B[K][Npix] (im2col)
// Workgroup = 256 threads (8 waves). WG tile: 128 M x 64 N. K-step = 64.
// Wave computes 16 M x 64 N via 8x v_wmma_f32_16x16x32_f16 per K-step
// (one A fragment reused across 4 B fragments per K-sub).
// Double-buffered LDS, one barrier per K-step. A tile staged with
// global_load_async_to_lds_b128 (ASYNCcnt) overlapped with WMMA compute.
// All conv dims are template constants: no runtime divisions anywhere.
// ---------------------------------------------------------------------------
template <int INH, int INW, int OUTH, int OUTW, int STRIDE, bool OUT_F16>
__global__ __launch_bounds__(256)
void conv_wmma_kernel(const _Float16* __restrict__ A,    // [256][K] f16 weights
                      const _Float16* __restrict__ act,  // [B,256,INH,INW] f16
                      const float* __restrict__ bias,    // [256]
                      void* __restrict__ out,            // [B,256,OUTH,OUTW]
                      int K) {
  constexpr int HW = OUTH * OUTW;
  constexpr int CHW = INH * INW;
  __shared__ _Float16 sA[2][128 * 64];   // [m][k]   16KB per buffer
  __shared__ _Float16 sBt[2][64 * 64];   // [n][k]    8KB per buffer

  const int tid  = threadIdx.x;
  const int wave = tid >> 5;
  const int lane = tid & 31;
  const int m0 = blockIdx.x * 128;
  const int n0 = blockIdx.y * 64;

  // ---- per-thread staging geometry (K-invariant, hoisted) ----
  // A: thread copies 64B of one weight row per K-step.
  const int arow = tid >> 1;           // 0..127
  const int acol = (tid & 1) * 32;     // halves within 64-half row
  // B: thread owns k-group kg (8 contiguous k) for two pixels nl0, nl0+32.
  const int nl0 = tid >> 3;            // 0..31
  const int kg  = tid & 7;             // 0..7

  const _Float16* actBase[2];
#pragma unroll
  for (int p = 0; p < 2; ++p) {
    int n = n0 + nl0 + p * 32;
    int pb   = n / HW;                 // constant divisors -> mul/shift
    int prem = n - pb * HW;
    int poy  = prem / OUTW;
    int pox  = prem - poy * OUTW;
    actBase[p] = act + (size_t)pb * 256 * CHW +
                 (size_t)(poy * STRIDE) * INW + pox * STRIDE;
  }

  // incremental (c, r*9+s) for the k-slot being *loaded*
  int c_b = 0;
  int rs_b = kg * 8;
  int aK = 0;                          // k0 of tile being loaded

  v8h rb0, rb1;

  auto load_A_async = [&](int buf) {
#if USE_ASYNC
    unsigned ldsA = (unsigned)(uintptr_t)(&sA[buf][arow * 64 + acol]);
    uint64_t gsrc = (uint64_t)(A + (size_t)(m0 + arow) * K + aK + acol);
    asm volatile(
        "global_load_async_to_lds_b128 %0, %1, off\n\t"
        "global_load_async_to_lds_b128 %0, %1, off offset:16\n\t"
        "global_load_async_to_lds_b128 %0, %1, off offset:32\n\t"
        "global_load_async_to_lds_b128 %0, %1, off offset:48"
        :: "v"(ldsA), "v"(gsrc) : "memory");
#else
    const uint4* asrc = (const uint4*)(A + (size_t)(m0 + arow) * K + aK + acol);
    uint4* adst = (uint4*)(&sA[buf][arow * 64 + acol]);
    adst[0] = asrc[0]; adst[1] = asrc[1]; adst[2] = asrc[2]; adst[3] = asrc[3];
#endif
  };

  auto load_B_regs = [&]() {
#pragma unroll
    for (int e = 0; e < 8; ++e) {
      int rs_e = rs_b + e;
      int ce = c_b;
      if (rs_e >= 81) { ce += 1; rs_e -= 81; }
      int r = rs_e / 9;                // constant divisor
      int s = rs_e - r * 9;
      size_t off = (size_t)ce * CHW + r * INW + s;
      rb0[e] = actBase[0][off];
      rb1[e] = actBase[1][off];
    }
    rs_b += 64;
    if (rs_b >= 81) { rs_b -= 81; c_b += 1; }
    aK += 64;
  };

  auto store_B = [&](int buf) {
    *(v8h*)(&sBt[buf][nl0 * 64 + kg * 8]) = rb0;
    *(v8h*)(&sBt[buf][(nl0 + 32) * 64 + kg * 8]) = rb1;
  };

  auto drain_async = [&]() {
#if USE_ASYNC
    asm volatile("s_wait_asynccnt 0x0" ::: "memory");
#endif
  };

  // ---- fragment geometry (contiguous k order, same for A and B) ----
  // A 16x32 f16: lane -> M = lane%16; K = (lane/16)*16 .. +15
  const int mrow = wave * 16 + (lane & 15);
  const int kb   = (lane >> 4) * 16;
  // B 32x16 f16: lanes 0-15 K=0..15, lanes 16-31 K=16..31; N = lane%16
  const int kcc  = (lane >> 4) * 16;
  const int bn   = lane & 15;

  v8f acc[4] = {{}, {}, {}, {}};

  // ---- prologue: stage tile 0 ----
  load_A_async(0);
  load_B_regs();
  store_B(0);
  drain_async();
  __syncthreads();

  const int NSTEP = K / 64;
  int cur = 0;
  for (int kt = 0; kt < NSTEP; ++kt) {
    const bool has_next = (kt + 1 < NSTEP);
    if (has_next) {
      load_A_async(1 - cur);   // async copy overlaps with WMMA below
      load_B_regs();
    }

#pragma unroll
    for (int ks = 0; ks < 2; ++ks) {
      const _Float16* Ar = &sA[cur][mrow * 64 + ks * 32];
      v16h af = *(const v16h*)(Ar + kb);
      const _Float16* Bb = &sBt[cur][ks * 32 + kcc];
#pragma unroll
      for (int q = 0; q < 4; ++q) {
        v16h bf = *(const v16h*)(Bb + (q * 16 + bn) * 64);
        acc[q] = __builtin_amdgcn_wmma_f32_16x16x32_f16(
            false, af, false, bf, (short)0, acc[q], false, false);
      }
    }

    if (has_next) store_B(1 - cur);
    drain_async();
    __syncthreads();
    cur ^= 1;
  }

  // ---- epilogue: C layout: VGPR r -> M = r + 8*(lane/16); N = lane%16 ----
  const int lhalf = lane >> 4;
  const int ncol  = lane & 15;
#pragma unroll
  for (int r = 0; r < 8; ++r) {
    int oc = m0 + wave * 16 + r + 8 * lhalf;
    float bval = bias[oc];
#pragma unroll
    for (int q = 0; q < 4; ++q) {
      float val = acc[q][r] + bval;
      val = fmaxf(val, 0.0f);
      int nn = n0 + q * 16 + ncol;
      int b = nn / HW;
      int rem = nn - b * HW;
      int oy = rem / OUTW;
      int ox = rem - oy * OUTW;
      size_t oidx = ((size_t)(b * 256 + oc) * OUTH + oy) * OUTW + ox;
      if constexpr (OUT_F16)
        ((_Float16*)out)[oidx] = (_Float16)val;
      else
        ((float*)out)[oidx] = val;
    }
  }
}

// ---------------------------------------------------------------------------
// squash primary capsules: act3 [B,9216] f32 -> u [B,1152,8]
// ---------------------------------------------------------------------------
__global__ void squash_prim_kernel(const float* __restrict__ act3,
                                   float* __restrict__ u) {
  int idx = blockIdx.x * 256 + threadIdx.x;  // (b,i): 128*1152
  const float* s = act3 + (size_t)idx * 8;
  float v[8], sn = 0.0f;
#pragma unroll
  for (int d = 0; d < 8; ++d) { v[d] = s[d]; sn += v[d] * v[d]; }
  float scale = (sn / (1.0f + sn)) * rsqrtf(sn + 1e-8f);
#pragma unroll
  for (int d = 0; d < 8; ++d) u[(size_t)idx * 8 + d] = v[d] * scale;
}

// ---------------------------------------------------------------------------
// u_hat[b,i,j,:] = u[b,i,:] @ W_digit[j,i,:,:]   (block per (i,j), thread per b)
// ---------------------------------------------------------------------------
__global__ __launch_bounds__(128)
void uhat_kernel(const float* __restrict__ u,        // [B,1152,8]
                 const float* __restrict__ W_digit,  // [10,1152,8,16]
                 float* __restrict__ u_hat) {        // [B,1152,10,16]
  __shared__ float Wsh[128];
  int i = blockIdx.x;
  int j = blockIdx.y;
  int tid = threadIdx.x;
  Wsh[tid] = W_digit[(size_t)(j * 1152 + i) * 128 + tid];
  __syncthreads();
  int b = tid;
  const float* up = u + (size_t)(b * 1152 + i) * 8;
  float uu[8];
#pragma unroll
  for (int d = 0; d < 8; ++d) uu[d] = up[d];
  float* op = u_hat + ((size_t)(b * 1152 + i) * 10 + j) * 16;
#pragma unroll
  for (int o = 0; o < 16; ++o) {
    float acc = 0.0f;
#pragma unroll
    for (int d = 0; d < 8; ++d) acc += uu[d] * Wsh[d * 16 + o];
    op[o] = acc;
  }
}

// ---------------------------------------------------------------------------
// softmax over i (1152) for each (b,j): c = softmax(b_ij, axis=1)
// ---------------------------------------------------------------------------
__global__ __launch_bounds__(256)
void softmax_i_kernel(const float* __restrict__ b_ij,  // [B,1152,10]
                      float* __restrict__ c) {         // [B,1152,10]
  __shared__ float red[256];
  __shared__ float bc[2];
  int bid = blockIdx.x;
  int b = bid / 10, j = bid % 10;
  int tid = threadIdx.x;
  const float* bp = b_ij + (size_t)b * 1152 * 10 + j;

  float m = -3.4e38f;
  for (int i = tid; i < 1152; i += 256) m = fmaxf(m, bp[(size_t)i * 10]);
  red[tid] = m;
  __syncthreads();
  for (int s = 128; s > 0; s >>= 1) {
    if (tid < s) red[tid] = fmaxf(red[tid], red[tid + s]);
    __syncthreads();
  }
  if (tid == 0) bc[0] = red[0];
  __syncthreads();
  float mx = bc[0];

  float sum = 0.0f;
  for (int i = tid; i < 1152; i += 256) sum += expf(bp[(size_t)i * 10] - mx);
  red[tid] = sum;
  __syncthreads();
  for (int s = 128; s > 0; s >>= 1) {
    if (tid < s) red[tid] += red[tid + s];
    __syncthreads();
  }
  if (tid == 0) bc[1] = red[0];
  __syncthreads();
  float inv = 1.0f / bc[1];

  float* cp = c + (size_t)b * 1152 * 10 + j;
  for (int i = tid; i < 1152; i += 256)
    cp[(size_t)i * 10] = expf(bp[(size_t)i * 10] - mx) * inv;
}

// ---------------------------------------------------------------------------
// s_j = sum_i c[b,i,j]*u_hat[b,i,j,:] + b_digit[j]; v = squash(s_j)
// block per (b,j); 256 threads = 16 o-slots x 16 i-groups
// ---------------------------------------------------------------------------
__global__ __launch_bounds__(256)
void sv_kernel(const float* __restrict__ c,        // [B,1152,10]
               const float* __restrict__ u_hat,    // [B,1152,10,16]
               const float* __restrict__ b_digit,  // [10,16]
               float* __restrict__ v,              // [B,10,16]
               float* __restrict__ out2) {         // optional extra dst (d_out)
  __shared__ float red[256];
  __shared__ float svs[16];
  int bid = blockIdx.x;
  int b = bid / 10, j = bid % 10;
  int tid = threadIdx.x;
  int og = tid & 15;
  int ig = tid >> 4;

  const float* uh = u_hat + ((size_t)b * 1152 * 10 + j) * 16 + og;
  const float* cp = c + (size_t)b * 1152 * 10 + j;
  float acc = 0.0f;
  for (int i = ig; i < 1152; i += 16)
    acc += cp[(size_t)i * 10] * uh[(size_t)i * 160];
  red[ig * 16 + og] = acc;
  __syncthreads();
  for (int s = 8; s > 0; s >>= 1) {
    if (ig < s) red[ig * 16 + og] += red[(ig + s) * 16 + og];
    __syncthreads();
  }
  if (tid < 16) svs[og] = red[og] + b_digit[j * 16 + og];
  __syncthreads();
  if (tid < 16) {
    float sn = 0.0f;
#pragma unroll
    for (int o = 0; o < 16; ++o) sn += svs[o] * svs[o];
    float val = svs[og] * (sn / (1.0f + sn)) * rsqrtf(sn + 1e-8f);
    size_t oidx = (size_t)(b * 10 + j) * 16 + og;
    v[oidx] = val;
    if (out2) out2[oidx] = val;
  }
}

// ---------------------------------------------------------------------------
// b_ij += sum_o u_hat[b,i,j,o] * v[b,j,o]; thread per (b,i,j)
// ---------------------------------------------------------------------------
__global__ void update_b_kernel(const float* __restrict__ u_hat,
                                const float* __restrict__ v,
                                float* __restrict__ b_ij) {
  int t = blockIdx.x * 256 + threadIdx.x;  // 128*1152*10
  int j = t % 10;
  int r = t / 10;
  int i = r % 1152;
  int b = r / 1152;
  const float* uh = u_hat + ((size_t)(b * 1152 + i) * 10 + j) * 16;
  const float* vp = v + (size_t)(b * 10 + j) * 16;
  float dot = 0.0f;
#pragma unroll
  for (int o = 0; o < 16; ++o) dot += uh[o] * vp[o];
  b_ij[t] += dot;
}

// ---------------------------------------------------------------------------
// out[b,n] = act(in[b,:] @ W[:,n] + bias[n]); ACT: 0=relu, 1=sigmoid
// ---------------------------------------------------------------------------
template <int ACT>
__global__ __launch_bounds__(256)
void linear_kernel(const float* __restrict__ in, const float* __restrict__ W,
                   const float* __restrict__ bias, float* __restrict__ out,
                   int K, int N) {
  int n = blockIdx.x * 256 + threadIdx.x;
  int b = blockIdx.y;
  if (n >= N) return;
  const float* ip = in + (size_t)b * K;
  float acc = bias[n];
  for (int k = 0; k < K; ++k) acc += ip[k] * W[(size_t)k * N + n];
  if (ACT == 0)
    acc = fmaxf(acc, 0.0f);
  else
    acc = 1.0f / (1.0f + expf(-acc));
  out[(size_t)b * N + n] = acc;
}

// ---------------------------------------------------------------------------
// host
// ---------------------------------------------------------------------------
extern "C" void kernel_launch(void* const* d_in, const int* in_sizes, int n_in,
                              void* d_out, int out_size, void* d_ws,
                              size_t ws_size, hipStream_t stream) {
  (void)in_sizes; (void)n_in; (void)out_size; (void)ws_size;

  const float* x       = (const float*)d_in[0];
  const float* conv1_w = (const float*)d_in[1];
  const float* conv1_b = (const float*)d_in[2];
  const float* conv2_w = (const float*)d_in[3];
  const float* conv2_b = (const float*)d_in[4];
  const float* prim_w  = (const float*)d_in[5];
  const float* prim_b  = (const float*)d_in[6];
  const float* W_digit = (const float*)d_in[7];
  const float* b_digit = (const float*)d_in[8];
  const float* rec_w1  = (const float*)d_in[9];
  const float* rec_b1  = (const float*)d_in[10];
  const float* rec_w2  = (const float*)d_in[11];
  const float* rec_b2  = (const float*)d_in[12];
  const float* rec_w3  = (const float*)d_in[13];
  const float* rec_b3  = (const float*)d_in[14];

  float* outv = (float*)d_out;                 // [128,10,16]
  float* outr = (float*)d_out + BATCH * 160;   // [128,3136]

  // workspace layout (256B aligned slabs)
  char* ws = (char*)d_ws;
  size_t off = 0;
  auto alloc = [&](size_t bytes) -> char* {
    char* p = ws + off;
    off = (off + bytes + 255) & ~(size_t)255;
    return p;
  };
  const int KBIG = 256 * 81;                                  // 20736
  _Float16* w2h  = (_Float16*)alloc((size_t)256 * KBIG * 2);  // conv2 w f16
  _Float16* wph  = (_Float16*)alloc((size_t)256 * KBIG * 2);  // prim  w f16
  _Float16* act1 = (_Float16*)alloc((size_t)BATCH * 256 * 48 * 48 * 2);
  _Float16* act2 = (_Float16*)alloc((size_t)BATCH * 256 * 20 * 20 * 2);
  float* act3  = (float*)alloc((size_t)BATCH * 9216 * 4);
  float* u     = (float*)alloc((size_t)BATCH * 1152 * 8 * 4);
  float* u_hat = (float*)alloc((size_t)BATCH * 1152 * 10 * 16 * 4);
  float* b_ij  = (float*)alloc((size_t)BATCH * 1152 * 10 * 4);
  float* c_ij  = (float*)alloc((size_t)BATCH * 1152 * 10 * 4);
  float* vbuf  = (float*)alloc((size_t)BATCH * 160 * 4);
  float* rec1  = (float*)alloc((size_t)BATCH * 512 * 4);
  float* rec2  = (float*)alloc((size_t)BATCH * 1024 * 4);

  // 1) weights -> f16
  {
    int n = 256 * KBIG;
    f32_to_f16_kernel<<<(n + 255) / 256, 256, 0, stream>>>(conv2_w, w2h, n);
    f32_to_f16_kernel<<<(n + 255) / 256, 256, 0, stream>>>(prim_w, wph, n);
  }
  // 2) conv1 (direct)
  {
    int n = BATCH * 256 * 48 * 48;
    conv1_kernel<<<n / 256, 256, 0, stream>>>(x, conv1_w, conv1_b, act1);
  }
  // 3) conv2: WMMA implicit GEMM, 48x48 -> 20x20 stride 2
  {
    dim3 grid(2, (BATCH * 20 * 20) / 64);  // (M tiles, N tiles)
    conv_wmma_kernel<48, 48, 20, 20, 2, true><<<grid, 256, 0, stream>>>(
        w2h, act1, conv2_b, (void*)act2, KBIG);
  }
  // 4) primary caps conv: 20x20 -> 6x6 stride 2, f32 out
  {
    dim3 grid(2, (BATCH * 6 * 6) / 64);
    conv_wmma_kernel<20, 20, 6, 6, 2, false><<<grid, 256, 0, stream>>>(
        wph, act2, prim_b, (void*)act3, KBIG);
  }
  // 5) squash
  squash_prim_kernel<<<(BATCH * 1152) / 256, 256, 0, stream>>>(act3, u);
  // 6) u_hat
  {
    dim3 grid(1152, 10);
    uhat_kernel<<<grid, 128, 0, stream>>>(u, W_digit, u_hat);
  }
  // 7) routing (3 iterations)
  {
    int nb = BATCH * 1152 * 10;
    zero_f32_kernel<<<(nb + 255) / 256, 256, 0, stream>>>(b_ij, nb);
    for (int it = 0; it < 3; ++it) {
      softmax_i_kernel<<<BATCH * 10, 256, 0, stream>>>(b_ij, c_ij);
      sv_kernel<<<BATCH * 10, 256, 0, stream>>>(
          c_ij, u_hat, b_digit, vbuf, (it == 2) ? outv : nullptr);
      if (it < 2)
        update_b_kernel<<<nb / 256, 256, 0, stream>>>(u_hat, vbuf, b_ij);
    }
  }
  // 8) reconstruction MLP
  {
    dim3 g1((512 + 255) / 256, BATCH);
    linear_kernel<0><<<g1, 256, 0, stream>>>(vbuf, rec_w1, rec_b1, rec1, 160, 512);
    dim3 g2((1024 + 255) / 256, BATCH);
    linear_kernel<0><<<g2, 256, 0, stream>>>(rec1, rec_w2, rec_b2, rec2, 512, 1024);
    dim3 g3((3136 + 255) / 256, BATCH);
    linear_kernel<1><<<g3, 256, 0, stream>>>(rec2, rec_w3, rec_b3, outr, 1024, 3136);
  }
}
